// GPT_46617575031424
// MI455X (gfx1250) — compile-verified, matrix operand order checked by reading, not run on hardware
//
#include <hip/hip_runtime.h>
#include <hip/hip_bf16.h>
#include <math.h>

// GPT-2 124M forward for gfx1250 (MI455X), wave32.
// - All GEMMs: v_wmma_f32_16x16x32_bf16, 128x128 block tile, 8 waves,
//   32x64 per wave (8 WMMA / wave / K-step).
// - Tile staging: global_load_async_to_lds_b128 (ASYNCcnt) with double
//   buffering; activations/weights pre-converted to bf16 by cvt_bf16.
// - Attention: flash-style streaming softmax, QK^T and PV through WMMA.
// Workspace requirement ~153 MB.

#define V_  50257
#define VP_ 50304           // V padded to multiple of 128 for lm_head tiles
#define E_  768
#define H_  12
#define L_  12
#define T_  1024
#define B_  2
#define D_  64
#define FF_ 3072

typedef __attribute__((ext_vector_type(16))) __bf16          v16bf;
typedef __attribute__((ext_vector_type(8)))  float           v8f;
typedef __attribute__((ext_vector_type(16))) unsigned short  u16x16;

// float -> bf16 bits, round-to-nearest-even.
__device__ __forceinline__ unsigned short f2bf(float f) {
  unsigned int u = __float_as_uint(f);
  unsigned int r = (u + 0x7FFFu + ((u >> 16) & 1u)) >> 16;
  return (unsigned short)r;
}

// 16-bit A/B fragment K-index for element e (0..15), lane-half hl (0/1).
// ISA 7.12.2: VGPR0..3 hold K=0..7 (lanes 0-15) / 8..15 (lanes 16-31);
// VGPR4..7 hold K=16..23 / 24..31; two 16-bit elements per VGPR.
__device__ __forceinline__ int wk(int e, int hl) {
  int v = e >> 1, p = e & 1;
  return ((v < 4) ? 0 : 16) + 2 * (v & 3) + p + hl * 8;
}

// Async global->LDS copy of 16 bytes (CDNA5, tracked by ASYNCcnt).
__device__ __forceinline__ void async_b128(unsigned lds_off, const void* g) {
  asm volatile("global_load_async_to_lds_b128 %0, %1, off"
               :: "v"(lds_off), "v"((unsigned long long)g) : "memory");
}
__device__ __forceinline__ void wait_async0() {
  asm volatile("s_wait_asynccnt 0x0" ::: "memory");
}
// Low 32 bits of a flat shared pointer == LDS byte offset (ISA 10.2).
__device__ __forceinline__ unsigned lds_off_of(const void* p) {
  return (unsigned)(size_t)p;
}

// ---------------------------------------------------------------------------
// fp32 -> bf16 bulk convert; indices >= nsrc write 0 (pad rows for lm_head).
// ---------------------------------------------------------------------------
__global__ __launch_bounds__(256) void cvt_bf16_kernel(
    const float* __restrict__ src, unsigned short* __restrict__ dst,
    long long n, long long nsrc) {
  long long i = (long long)blockIdx.x * 256 + threadIdx.x;
  long long stride = (long long)gridDim.x * 256;
  for (; i < n; i += stride)
    dst[i] = (i < nsrc) ? f2bf(src[i]) : (unsigned short)0;
}

// ---------------------------------------------------------------------------
// Embedding: x[b,t,:] = wte[idx[b,t],:] + wpe[t,:]
// ---------------------------------------------------------------------------
__global__ __launch_bounds__(256) void embed_kernel(
    const int* __restrict__ idx, const float* __restrict__ wte,
    const float* __restrict__ wpe, float* __restrict__ x) {
  int row = blockIdx.x;
  int t   = row % T_;
  int tok = idx[row];
  int tid = threadIdx.x;
#pragma unroll
  for (int j = 0; j < 3; ++j) {
    int e = tid + j * 256;
    x[(size_t)row * E_ + e] = wte[(size_t)tok * E_ + e] + wpe[(size_t)t * E_ + e];
  }
}

// ---------------------------------------------------------------------------
// LayerNorm: one 256-thread block per row of E=768.
// ---------------------------------------------------------------------------
__global__ __launch_bounds__(256) void ln_kernel(
    const float* __restrict__ x, const float* __restrict__ w,
    const float* __restrict__ b, float* __restrict__ out) {
  __shared__ float red[2][8];
  int row = blockIdx.x;
  int tid = threadIdx.x;
  const float* xp = x + (size_t)row * E_;
  float v[3];
  float s = 0.f, s2 = 0.f;
#pragma unroll
  for (int j = 0; j < 3; ++j) {
    v[j] = xp[tid + j * 256];
    s += v[j]; s2 += v[j] * v[j];
  }
#pragma unroll
  for (int m = 16; m >= 1; m >>= 1) {
    s  += __shfl_xor(s,  m, 32);
    s2 += __shfl_xor(s2, m, 32);
  }
  int lane = tid & 31, wid = tid >> 5;
  if (lane == 0) { red[0][wid] = s; red[1][wid] = s2; }
  __syncthreads();
  float ts = 0.f, ts2 = 0.f;
#pragma unroll
  for (int i = 0; i < 8; ++i) { ts += red[0][i]; ts2 += red[1][i]; }
  float mu  = ts  * (1.0f / E_);
  float var = ts2 * (1.0f / E_) - mu * mu;
  float rs  = rsqrtf(var + 1e-5f);
  float* op = out + (size_t)row * E_;
#pragma unroll
  for (int j = 0; j < 3; ++j) {
    int e = tid + j * 256;
    op[e] = (v[j] - mu) * rs * w[e] + b[e];
  }
}

// ---------------------------------------------------------------------------
// WMMA GEMM on bf16 operands: C = epi(A[M,K] @ W + bias [+ res]), fp32 out.
// WT=false: W is [K,N] bf16. WT=true: W is [N,K] bf16 (lm_head over wte).
// 128x128 block tile, 8 waves (4 row-groups x 2 col-groups), each wave a
// 32x64 sub-tile = 2 A-frags x 4 B-frags = 8 WMMAs per 32-deep K-step.
// Staging: double-buffered async global->LDS b128 copies (no VGPR pass).
// EPI: 0 = bias, 1 = bias+residual, 2 = bias+tanh-GELU.
// ---------------------------------------------------------------------------
template <int EPI, bool WT>
__global__ __launch_bounds__(256) void gemm_wmma(
    const unsigned short* __restrict__ A, const unsigned short* __restrict__ W,
    const float* __restrict__ bias, const float* __restrict__ res,
    float* __restrict__ C, int M, int N, int K) {
  constexpr int BROWS  = WT ? 128 : 32;   // LDS W-tile rows
  constexpr int BPITCH = WT ? 40 : 136;   // padded pitch (u16), 16B multiple
  __shared__ __align__(16) unsigned short sA[2][128][40];
  __shared__ __align__(16) unsigned short sB[2][BROWS][BPITCH];

  const int tid  = threadIdx.x;
  const int lane = tid & 31;
  const int w    = tid >> 5;
  const int wr   = w & 3, wc = w >> 2;
  const int m0   = blockIdx.y * 128;
  const int n0   = blockIdx.x * 128;
  const int hl   = lane >> 4, l15 = lane & 15;
  const unsigned sA_base = lds_off_of(&sA[0][0][0]);
  const unsigned sB_base = lds_off_of(&sB[0][0][0]);

  auto issue = [&](int buf, int k0) {
    // A tile [128 x 32] bf16: 512 x b128 chunks, 2 per thread.
#pragma unroll
    for (int i = 0; i < 2; ++i) {
      int c = tid + i * 256;
      int row = c >> 2, cc = (c & 3) * 8;
      const unsigned short* g = A + (size_t)(m0 + row) * K + k0 + cc;
      async_b128(sA_base + (unsigned)(((buf * 128 + row) * 40 + cc) * 2), g);
    }
    // W tile: [32 k x 128 n] (WT=false) or [128 n x 32 k] (WT=true).
#pragma unroll
    for (int i = 0; i < 2; ++i) {
      int c = tid + i * 256;
      int row, cc;
      const unsigned short* g;
      if (WT) {
        row = c >> 2; cc = (c & 3) * 8;
        g = W + (size_t)(n0 + row) * K + k0 + cc;
      } else {
        row = c >> 4; cc = (c & 15) * 8;
        g = W + (size_t)(k0 + row) * N + n0 + cc;
      }
      async_b128(sB_base + (unsigned)(((buf * BROWS + row) * BPITCH + cc) * 2), g);
    }
  };

  v8f acc[2][4];
#pragma unroll
  for (int ti = 0; ti < 2; ++ti)
#pragma unroll
    for (int tj = 0; tj < 4; ++tj)
#pragma unroll
      for (int r = 0; r < 8; ++r) acc[ti][tj][r] = 0.f;

  issue(0, 0);
  wait_async0();
  __syncthreads();

  const int nk = K / 32;
  for (int kb = 0; kb < nk; ++kb) {
    const int cur = kb & 1;
    if (kb + 1 < nk) issue(1 - cur, (kb + 1) * 32);

    u16x16 au0, au1;
#pragma unroll
    for (int e = 0; e < 16; ++e) {
      int k = wk(e, hl);
      au0[e] = sA[cur][wr * 32 + l15][k];
      au1[e] = sA[cur][wr * 32 + 16 + l15][k];
    }
    v16bf a0 = __builtin_bit_cast(v16bf, au0);
    v16bf a1 = __builtin_bit_cast(v16bf, au1);
#pragma unroll
    for (int tj = 0; tj < 4; ++tj) {
      u16x16 bu;
#pragma unroll
      for (int e = 0; e < 16; ++e) {
        int k = wk(e, hl);
        int n = wc * 64 + tj * 16 + l15;
        bu[e] = WT ? sB[cur][n][k] : sB[cur][k][n];
      }
      v16bf b = __builtin_bit_cast(v16bf, bu);
      acc[0][tj] = __builtin_amdgcn_wmma_f32_16x16x32_bf16(
          false, a0, false, b, (short)0, acc[0][tj], false, false);
      acc[1][tj] = __builtin_amdgcn_wmma_f32_16x16x32_bf16(
          false, a1, false, b, (short)0, acc[1][tj], false, false);
    }
    wait_async0();
    __syncthreads();
  }

  // Epilogue (C/D layout: VGPR r -> row r + 8*hl, lane l15 -> col).
#pragma unroll
  for (int ti = 0; ti < 2; ++ti)
#pragma unroll
    for (int r = 0; r < 8; ++r) {
      int m = m0 + wr * 32 + ti * 16 + r + 8 * hl;
#pragma unroll
      for (int tj = 0; tj < 4; ++tj) {
        int n = n0 + wc * 64 + tj * 16 + l15;
        if (n < N) {
          float v = acc[ti][tj][r];
          if (bias) v += bias[n];
          if (EPI == 1) v += res[(size_t)m * N + n];
          if (EPI == 2) {
            float u = v;
            v = 0.5f * u *
                (1.0f + tanhf(0.7978845608028654f * (u + 0.044715f * u * u * u)));
          }
          C[(size_t)m * N + n] = v;
        }
      }
    }
}

// ---------------------------------------------------------------------------
// Flash attention: one wave32 per (b, h, 16-query tile). Streams 32-key
// blocks; Q@K^T and P@V via WMMA; online softmax; causal mask.
// ---------------------------------------------------------------------------
__global__ __launch_bounds__(32) void attn_wmma(
    const float* __restrict__ qkv, float* __restrict__ y) {
  __shared__ unsigned short sQ[16][66];
  __shared__ unsigned short sK[32][66];
  __shared__ unsigned short sV[32][66];
  __shared__ unsigned short sP[16][34];
  const int lane = threadIdx.x;
  const int hl = lane >> 4, l15 = lane & 15;
  const int qt = blockIdx.x;
  const int h  = blockIdx.y;
  const int b  = blockIdx.z;
  const size_t rs3 = 3 * E_;
  const float scale = 0.125f;  // 1/sqrt(D=64)

  {
    int row = lane >> 1;
    int c0  = (lane & 1) * 32;
    const float* qp =
        qkv + ((size_t)(b * T_) + qt * 16 + row) * rs3 + h * D_ + c0;
    for (int j = 0; j < 32; ++j) sQ[row][c0 + j] = f2bf(qp[j]);
  }
  __syncthreads();
  u16x16 q0u, q1u;
#pragma unroll
  for (int e = 0; e < 16; ++e) {
    int k  = wk(e, hl);
    q0u[e] = sQ[l15][k];
    q1u[e] = sQ[l15][32 + k];
  }
  v16bf aq0 = __builtin_bit_cast(v16bf, q0u);
  v16bf aq1 = __builtin_bit_cast(v16bf, q1u);

  float rmax[8], rsum[8];
  v8f yacc[4];
#pragma unroll
  for (int r = 0; r < 8; ++r) {
    rmax[r] = -1e30f; rsum[r] = 0.f;
#pragma unroll
    for (int c = 0; c < 4; ++c) yacc[c][r] = 0.f;
  }

  const int qmax = qt * 16 + 15;
  const int nblk = (qmax + 32) / 32;

  for (int kb = 0; kb < nblk; ++kb) {
    __syncthreads();
    {
      const float* kp =
          qkv + ((size_t)(b * T_) + kb * 32 + lane) * rs3 + E_ + h * D_;
      const float* vp = kp + E_;
      for (int j = 0; j < 64; ++j) {
        sK[lane][j] = f2bf(kp[j]);
        sV[lane][j] = f2bf(vp[j]);
      }
    }
    __syncthreads();

    v8f s0, s1;
#pragma unroll
    for (int r = 0; r < 8; ++r) { s0[r] = 0.f; s1[r] = 0.f; }
    u16x16 bt;
#pragma unroll
    for (int e = 0; e < 16; ++e) bt[e] = sK[l15][wk(e, hl)];
    s0 = __builtin_amdgcn_wmma_f32_16x16x32_bf16(
        false, aq0, false, __builtin_bit_cast(v16bf, bt), (short)0, s0, false, false);
#pragma unroll
    for (int e = 0; e < 16; ++e) bt[e] = sK[l15][32 + wk(e, hl)];
    s0 = __builtin_amdgcn_wmma_f32_16x16x32_bf16(
        false, aq1, false, __builtin_bit_cast(v16bf, bt), (short)0, s0, false, false);
#pragma unroll
    for (int e = 0; e < 16; ++e) bt[e] = sK[16 + l15][wk(e, hl)];
    s1 = __builtin_amdgcn_wmma_f32_16x16x32_bf16(
        false, aq0, false, __builtin_bit_cast(v16bf, bt), (short)0, s1, false, false);
#pragma unroll
    for (int e = 0; e < 16; ++e) bt[e] = sK[16 + l15][32 + wk(e, hl)];
    s1 = __builtin_amdgcn_wmma_f32_16x16x32_bf16(
        false, aq1, false, __builtin_bit_cast(v16bf, bt), (short)0, s1, false, false);

#pragma unroll
    for (int r = 0; r < 8; ++r) {
      int qrow = qt * 16 + r + 8 * hl;
      int key0 = kb * 32 + l15;
      float v0 = (key0 <= qrow)      ? s0[r] * scale : -1e30f;
      float v1 = (key0 + 16 <= qrow) ? s1[r] * scale : -1e30f;
      float mx = fmaxf(v0, v1);
#pragma unroll
      for (int m = 8; m >= 1; m >>= 1) mx = fmaxf(mx, __shfl_xor(mx, m, 32));
      float newm = fmaxf(rmax[r], mx);
      float sf = __expf(rmax[r] - newm);
      float p0 = __expf(v0 - newm);
      float p1 = __expf(v1 - newm);
      float ps = p0 + p1;
#pragma unroll
      for (int m = 8; m >= 1; m >>= 1) ps += __shfl_xor(ps, m, 32);
      rsum[r] = rsum[r] * sf + ps;
      rmax[r] = newm;
#pragma unroll
      for (int c = 0; c < 4; ++c) yacc[c][r] *= sf;
      sP[r + 8 * hl][l15]      = f2bf(p0);
      sP[r + 8 * hl][16 + l15] = f2bf(p1);
    }
    __syncthreads();

    u16x16 pu;
#pragma unroll
    for (int e = 0; e < 16; ++e) pu[e] = sP[l15][wk(e, hl)];
    v16bf ap = __builtin_bit_cast(v16bf, pu);
#pragma unroll
    for (int c = 0; c < 4; ++c) {
      u16x16 bv;
#pragma unroll
      for (int e = 0; e < 16; ++e) bv[e] = sV[wk(e, hl)][c * 16 + l15];
      yacc[c] = __builtin_amdgcn_wmma_f32_16x16x32_bf16(
          false, ap, false, __builtin_bit_cast(v16bf, bv), (short)0, yacc[c],
          false, false);
    }
  }

#pragma unroll
  for (int r = 0; r < 8; ++r) {
    float inv = 1.0f / rsum[r];
    int m = qt * 16 + r + 8 * hl;
#pragma unroll
    for (int c = 0; c < 4; ++c) {
      y[((size_t)(b * T_) + m) * E_ + h * D_ + c * 16 + l15] = yacc[c][r] * inv;
    }
  }
}

// ---------------------------------------------------------------------------
// Host orchestration (graph-capture safe: only kernel launches on `stream`).
// ---------------------------------------------------------------------------
extern "C" void kernel_launch(void* const* d_in, const int* in_sizes, int n_in,
                              void* d_out, int out_size, void* d_ws,
                              size_t ws_size, hipStream_t stream) {
  (void)in_sizes; (void)n_in; (void)out_size; (void)ws_size;
  const int*   idx    = (const int*)  d_in[0];
  const float* wte    = (const float*)d_in[1];
  const float* wpe    = (const float*)d_in[2];
  const float* ln1_w  = (const float*)d_in[3];
  const float* ln1_b  = (const float*)d_in[4];
  const float* attn_w = (const float*)d_in[5];
  const float* attn_b = (const float*)d_in[6];
  const float* proj_w = (const float*)d_in[7];
  const float* proj_b = (const float*)d_in[8];
  const float* ln2_w  = (const float*)d_in[9];
  const float* ln2_b  = (const float*)d_in[10];
  const float* fc_w   = (const float*)d_in[11];
  const float* fc_b   = (const float*)d_in[12];
  const float* fcp_w  = (const float*)d_in[13];
  const float* fcp_b  = (const float*)d_in[14];
  const float* lnf_w  = (const float*)d_in[15];
  const float* lnf_b  = (const float*)d_in[16];

  const int M = B_ * T_;  // 2048
  float* x   = (float*)d_ws;
  float* h   = x   + (size_t)M * E_;
  float* qkv = h   + (size_t)M * E_;
  float* yb  = qkv + (size_t)M * 3 * E_;
  float* ff  = yb  + (size_t)M * E_;
  unsigned short* abf = (unsigned short*)(ff + (size_t)M * FF_);
  unsigned short* wbf = abf + (size_t)M * FF_;   // sized VP_*E_ (lm_head max)

  auto cvt = [&](const float* s, unsigned short* d, long long n, long long ns) {
    int blocks = (int)((n + 1023) / 1024);
    cvt_bf16_kernel<<<blocks, 256, 0, stream>>>(s, d, n, ns);
  };

  embed_kernel<<<M, 256, 0, stream>>>(idx, wte, wpe, x);

  for (int l = 0; l < L_; ++l) {
    // --- attention ---
    ln_kernel<<<M, 256, 0, stream>>>(x, ln1_w + l * E_, ln1_b + l * E_, h);
    cvt(h, abf, (long long)M * E_, (long long)M * E_);
    cvt(attn_w + (size_t)l * E_ * 3 * E_, wbf, (long long)E_ * 3 * E_,
        (long long)E_ * 3 * E_);
    gemm_wmma<0, false><<<dim3((3 * E_) / 128, M / 128), 256, 0, stream>>>(
        abf, wbf, attn_b + (size_t)l * 3 * E_, nullptr, qkv, M, 3 * E_, E_);
    attn_wmma<<<dim3(T_ / 16, H_, B_), 32, 0, stream>>>(qkv, yb);
    cvt(yb, abf, (long long)M * E_, (long long)M * E_);
    cvt(proj_w + (size_t)l * E_ * E_, wbf, (long long)E_ * E_,
        (long long)E_ * E_);
    gemm_wmma<1, false><<<dim3(E_ / 128, M / 128), 256, 0, stream>>>(
        abf, wbf, proj_b + (size_t)l * E_, x, x, M, E_, E_);
    // --- MLP ---
    ln_kernel<<<M, 256, 0, stream>>>(x, ln2_w + l * E_, ln2_b + l * E_, h);
    cvt(h, abf, (long long)M * E_, (long long)M * E_);
    cvt(fc_w + (size_t)l * E_ * FF_, wbf, (long long)E_ * FF_,
        (long long)E_ * FF_);
    gemm_wmma<2, false><<<dim3(FF_ / 128, M / 128), 256, 0, stream>>>(
        abf, wbf, fc_b + (size_t)l * FF_, nullptr, ff, M, FF_, E_);
    cvt(ff, abf, (long long)M * FF_, (long long)M * FF_);
    cvt(fcp_w + (size_t)l * FF_ * E_, wbf, (long long)FF_ * E_,
        (long long)FF_ * E_);
    gemm_wmma<1, false><<<dim3(E_ / 128, M / 128), 256, 0, stream>>>(
        abf, wbf, fcp_b + (size_t)l * E_, x, x, M, E_, FF_);
  }

  ln_kernel<<<M, 256, 0, stream>>>(x, lnf_w, lnf_b, h);
  cvt(h, abf, (long long)M * E_, (long long)M * E_);
  // wte [V,E] -> bf16 padded to VP_ rows (zero tail) for guard-free tiles.
  cvt(wte, wbf, (long long)VP_ * E_, (long long)V_ * E_);
  gemm_wmma<0, true><<<dim3(VP_ / 128, M / 128), 256, 0, stream>>>(
      abf, wbf, nullptr, nullptr, (float*)d_out, M, V_, E_);
}
